// MacheteLayer_29832842838090
// MI455X (gfx1250) — compile-verified, matrix operand order checked by reading, not run on hardware
//
#include <hip/hip_runtime.h>

// ---------------------------------------------------------------------------
// W4A16-style group-dequant GEMM for MI455X (gfx1250, wave32, WMMA).
//   a      : [64, 4096]  f32
//   b_q    : [4096, 14336] int32 codes in [0,16), zero-point 8
//   scales : [32, 14336] f32  (per 128-row K-group, per column)
//   out    : [64, 14336] f32 = a @ ((b_q - 8) * scales_expanded)
//
// Memory-bound on b_q (235 MB @ 23.3 TB/s ~ 10us floor). Required compute
// rate at that floor ~750 TFLOPS => v_wmma_f32_16x16x32_f16 matrix pipes.
// Dequant uses fma(cvt(c), s, -8s) + v_cvt_pk_rtz_f16_f32 to keep VALU
// cost ~2.5 ops/element so the stream stays HBM-bound, not VALU-bound.
// ---------------------------------------------------------------------------

#define MM   64
#define KK   4096
#define NN   14336
#define KCHUNK 32               // K per pipeline stage (one WMMA K-depth)
#define NTILE  64               // N columns per workgroup
#define NWAVES 4                // 128 threads per workgroup
#define NCHUNKS (KK / KCHUNK)   // 128
#define NGROUPCHUNKS 4          // 128-row quant group = 4 chunks

typedef __attribute__((ext_vector_type(16))) _Float16 v16h;
typedef __attribute__((ext_vector_type(8)))  float    v8f;
typedef __attribute__((ext_vector_type(2)))  __fp16   fp16x2;  // cvt_pkrtz type

// dequant pair -> packed f16 dword stored to LDS (v_cvt_pk_rtz_f16_f32)
__device__ __forceinline__ void store_pk(void* p, float x, float y) {
    *(fp16x2*)p = __builtin_amdgcn_cvt_pkrtz(x, y);
}

// ---------------------------------------------------------------------------
// Kernel 1: repack A (f32) into per-lane WMMA A-fragment order (f16).
// Layout: apack[kc][mt][lane][16 f16], 32B contiguous per (kc,mt,lane).
//   lane<16 : M = mt*16+lane,     K = k0 + {0..7, 16..23}
//   lane>=16: M = mt*16+lane-16,  K = k0 + {8..15, 24..31}
// Total: 128 * 4 * 32 * 16 f16 = 512 KB (lives in d_ws, L2-resident).
// ---------------------------------------------------------------------------
__global__ void pack_a_kernel(const float* __restrict__ a,
                              _Float16* __restrict__ apack) {
    const int idx = blockIdx.x * blockDim.x + threadIdx.x;   // 0..16383
    if (idx >= NCHUNKS * 4 * 32) return;
    const int lane = idx & 31;
    const int mt   = (idx >> 5) & 3;
    const int kc   = idx >> 7;

    const int m     = mt * 16 + (lane & 15);
    const int khalf = (lane >> 4) * 8;
    const int k0    = kc * KCHUNK;

    _Float16 vals[16];
#pragma unroll
    for (int j = 0; j < 16; ++j) {
        const int k = k0 + khalf + (j & 7) + ((j >> 3) * 16);
        vals[j] = (_Float16)a[(size_t)m * KK + k];
    }
    *(v16h*)(apack + (size_t)idx * 16) = *(const v16h*)vals;
}

// ---------------------------------------------------------------------------
// Kernel 2: dequant + GEMM via WMMA.
// ---------------------------------------------------------------------------
__global__ void __launch_bounds__(NWAVES * 32)
dequant_gemm_kernel(const _Float16* __restrict__ apack,
                    const int*      __restrict__ bq,
                    const float*    __restrict__ scales,
                    float*          __restrict__ out) {
    // LDS: double-buffered dequantized B tile in B-fragment byte order.
    // Per column n (0..63): 64 bytes = 32 f16 arranged as
    //   bytes  0..15 : K k0+0 .. k0+7      (g0)
    //   bytes 16..31 : K k0+16.. k0+23     (g2)
    //   bytes 32..47 : K k0+8 .. k0+15     (g1)
    //   bytes 48..63 : K k0+24.. k0+31     (g3)
    // so lane<16 reads [n][0..31], lane>=16 reads [n][32..63] as its v16h.
    __shared__ __align__(64) unsigned char ldsb[2][NTILE][64];

    const int t    = threadIdx.x;        // 0..127
    const int lane = t & 31;
    const int wave = t >> 5;             // 0..3
    const int n_wg = blockIdx.x * NTILE; // 224 workgroups

    // ---- staging role: each thread owns 4 fixed columns, 2x2 K rows/iter ---
    const int n4 = (t & 15) * 4;         // column quad within tile
    const int kp = (t >> 4) * 2;         // even k row pair base (0..14)

    v8f acc[4] = {};                     // 4 M-tiles x 16x16 f32

    // Scales for this thread's 4 columns (group 0 first) and folded
    // zero-point term ns = -8*s, so dequant is a single fma per element.
    float4 sv = *(const float4*)(scales + n_wg + n4);
    float4 nsv = make_float4(-8.0f * sv.x, -8.0f * sv.y,
                             -8.0f * sv.z, -8.0f * sv.w);

    // ---- stage one K-chunk into LDS buffer `buf` ----
    auto stage = [&](int kc, int buf, const float4& s, const float4& ns) {
        const int k0 = kc * KCHUNK;
#pragma unroll
        for (int i = 0; i < 2; ++i) {
            const int kl = kp + i * 16;                       // even, 0..30
            const size_t row = (size_t)(k0 + kl) * NN + n_wg + n4;
            const int4 c0 = *(const int4*)(bq + row);         // K = k0+kl
            const int4 c1 = *(const int4*)(bq + row + NN);    // K = k0+kl+1
            if (i == 0 && kc + 2 < NCHUNKS) {
                // prefetch b_q two chunks ahead (global_prefetch_b8)
                __builtin_prefetch(bq + row + (size_t)(2 * KCHUNK) * NN, 0, 1);
            }
            const int g = kl >> 3;                            // 0..3
            unsigned char* base = &ldsb[buf][0][0]
                                + (((g & 1) << 5) | (((g >> 1) & 1) << 4))
                                + ((kl & 7) << 1);
            // dequant: w = fma(cvt(c), s, -8s); pack via v_cvt_pk_rtz_f16_f32
            store_pk(base + (size_t)(n4 + 0) * 64,
                     fmaf((float)c0.x, s.x, ns.x), fmaf((float)c1.x, s.x, ns.x));
            store_pk(base + (size_t)(n4 + 1) * 64,
                     fmaf((float)c0.y, s.y, ns.y), fmaf((float)c1.y, s.y, ns.y));
            store_pk(base + (size_t)(n4 + 2) * 64,
                     fmaf((float)c0.z, s.z, ns.z), fmaf((float)c1.z, s.z, ns.z));
            store_pk(base + (size_t)(n4 + 3) * 64,
                     fmaf((float)c0.w, s.w, ns.w), fmaf((float)c1.w, s.w, ns.w));
        }
    };

    stage(0, 0, sv, nsv);
    __syncthreads();

    // B-fragment LDS read address for this lane (32B, v16h).
    const unsigned char* bfrag_addr =
        &ldsb[0][wave * 16 + (lane & 15)][(lane >> 4) * 32];
    const size_t lds_buf_stride = (size_t)NTILE * 64;

    for (int kc = 0; kc < NCHUNKS; ++kc) {
        const int cur = kc & 1;

        // Stage next chunk into the other buffer while computing this one.
        if (kc + 1 < NCHUNKS) {
            if (((kc + 1) & (NGROUPCHUNKS - 1)) == 0) {
                const int grp = (kc + 1) >> 2;
                sv = *(const float4*)(scales + (size_t)grp * NN + n_wg + n4);
                nsv = make_float4(-8.0f * sv.x, -8.0f * sv.y,
                                  -8.0f * sv.z, -8.0f * sv.w);
            }
            stage(kc + 1, cur ^ 1, sv, nsv);
        }

        // ---- compute: one B fragment, 4 A fragments, 4 WMMAs ----
        const v16h bfrag =
            *(const v16h*)(bfrag_addr + (size_t)cur * lds_buf_stride);
        const _Float16* ab = apack + ((size_t)(kc * 4) * 32 + lane) * 16;
#pragma unroll
        for (int mt = 0; mt < 4; ++mt) {
            const v16h afrag = *(const v16h*)(ab + (size_t)mt * 32 * 16);
            acc[mt] = __builtin_amdgcn_wmma_f32_16x16x32_f16(
                false, afrag, false, bfrag, (short)0, acc[mt], false, false);
        }
        __syncthreads();
    }

    // ---- epilogue: C layout -> out[m][n], each element written once ----
    const int nb  = n_wg + wave * 16 + (lane & 15);
    const int mhi = (lane >> 4) * 8;
#pragma unroll
    for (int mt = 0; mt < 4; ++mt) {
#pragma unroll
        for (int r = 0; r < 8; ++r) {
            out[(size_t)(mt * 16 + mhi + r) * NN + nb] = acc[mt][r];
        }
    }
}

// ---------------------------------------------------------------------------
// Host launcher.
// ---------------------------------------------------------------------------
extern "C" void kernel_launch(void* const* d_in, const int* in_sizes, int n_in,
                              void* d_out, int out_size, void* d_ws, size_t ws_size,
                              hipStream_t stream) {
    const float* a      = (const float*)d_in[0];
    const int*   bq     = (const int*)d_in[1];
    const float* scales = (const float*)d_in[2];
    float*       out    = (float*)d_out;

    _Float16* apack = (_Float16*)d_ws;   // 512 KB scratch for packed A

    // Repack A into WMMA fragment order (tiny: 16384 threads).
    pack_a_kernel<<<64, 256, 0, stream>>>(a, apack);

    // Main dequant GEMM: 14336/64 = 224 workgroups x 4 waves.
    dequant_gemm_kernel<<<NN / NTILE, NWAVES * 32, 0, stream>>>(
        apack, bq, scales, out);
}
